// MixerBlock_8400956031057
// MI455X (gfx1250) — matrix operand order repaired and checked by reading, not compiled, not run on hardware
//
#include <hip/hip_runtime.h>
#include <hip/hip_bf16.h>

// ---------------------------------------------------------------------------
// MixerBlock for MI455X (gfx1250): bf16 WMMA GEMMs with f32 accumulation,
// double-buffered async global->LDS staging (ASYNCcnt path).
// B=2, S=4096, E=1024, DF=4096.
// ---------------------------------------------------------------------------

typedef __attribute__((ext_vector_type(16))) __bf16 v16bf;
typedef __attribute__((ext_vector_type(8)))  __bf16 v8bf;
typedef __attribute__((ext_vector_type(8)))  float  v8f;

#define BDIM   2
#define SDIM   4096
#define EDIM   1024
#define DFDIM  4096
#define MTOT   (BDIM * SDIM)   // 8192 rows

#define BM 128   // block rows (8 waves x 16)
#define BN 64    // block cols
#define BK 32    // K step (wmma_f32_16x16x32_bf16)

__device__ __forceinline__ unsigned short f32_to_bf16_bits(float f) {
    unsigned int u = __builtin_bit_cast(unsigned int, f);
    u += 0x7FFFu + ((u >> 16) & 1u);   // round-to-nearest-even
    return (unsigned short)(u >> 16);
}

// Async direct global->LDS copy of 16 bytes (gfx1250 ASYNCcnt path).
__device__ __forceinline__ void async_copy_b128(unsigned short* lds_dst,
                                                const unsigned short* gsrc) {
    unsigned ldsa = (unsigned)(size_t)lds_dst;            // low 32b = LDS offset
    unsigned long long ga = (unsigned long long)(size_t)gsrc;
    asm volatile("global_load_async_to_lds_b128 %0, %1, off"
                 :: "v"(ldsa), "v"(ga) : "memory");
}
__device__ __forceinline__ void wait_async() {
    asm volatile("s_wait_asynccnt 0x0" ::: "memory");
}

// ---------------------------------------------------------------------------
// Transpose + convert fp32 [K][N] -> bf16 [N][K]
// ---------------------------------------------------------------------------
__global__ void convT_kernel(const float* __restrict__ W,
                             unsigned short* __restrict__ WT,
                             int K_, int N_) {
    size_t i = (size_t)blockIdx.x * 256 + threadIdx.x;
    size_t total = (size_t)K_ * N_;
    if (i < total) {
        int n = (int)(i % N_);
        int k = (int)(i / N_);
        WT[(size_t)n * K_ + k] = f32_to_bf16_bits(W[i]);
    }
}

// ---------------------------------------------------------------------------
// Row LayerNorm over E=1024, output bf16 (optionally transposed [E][S]).
// ---------------------------------------------------------------------------
__global__ void ln_rows_kernel(const float* __restrict__ x,
                               const float* __restrict__ g,
                               const float* __restrict__ bvec,
                               unsigned short* __restrict__ out,
                               int S_, int transposed) {
    __shared__ float sbuf[256];
    const int row = blockIdx.x;
    const int tid = threadIdx.x;
    const float4 v4 = ((const float4*)(x + (size_t)row * EDIM))[tid];
    float vv[4] = {v4.x, v4.y, v4.z, v4.w};

    sbuf[tid] = vv[0] + vv[1] + vv[2] + vv[3]; __syncthreads();
    for (int o = 128; o > 0; o >>= 1) {
        if (tid < o) sbuf[tid] += sbuf[tid + o];
        __syncthreads();
    }
    const float mean = sbuf[0] * (1.0f / (float)EDIM);
    __syncthreads();
    float d0 = vv[0]-mean, d1 = vv[1]-mean, d2 = vv[2]-mean, d3 = vv[3]-mean;
    sbuf[tid] = d0*d0 + d1*d1 + d2*d2 + d3*d3; __syncthreads();
    for (int o = 128; o > 0; o >>= 1) {
        if (tid < o) sbuf[tid] += sbuf[tid + o];
        __syncthreads();
    }
    const float rstd = rsqrtf(sbuf[0] * (1.0f / (float)EDIM) + 1e-5f);

    const int bidx = row / S_;
    const int sidx = row % S_;
    #pragma unroll
    for (int j = 0; j < 4; ++j) {
        const int e = tid * 4 + j;
        const float y = (vv[j] - mean) * rstd * g[e] + bvec[e];
        const unsigned short h = f32_to_bf16_bits(y);
        if (transposed)
            out[((size_t)bidx * EDIM + e) * S_ + sidx] = h;
        else
            out[(size_t)row * EDIM + e] = h;
    }
}

// ---------------------------------------------------------------------------
// Per-wave 16x64 WMMA strip on one staged 64x32 B tile.
// ---------------------------------------------------------------------------
__device__ __forceinline__ void wmma_step(const unsigned short* Arow, int k0,
                                          const unsigned short* bt,
                                          int lane, v8f acc[4]) {
    const int kbase = (lane < 16) ? 0 : 16;
    const int bsel  = (lane < 16) ? 0 : 16;
    const v8bf alo = *(const v8bf*)(Arow + k0 + kbase);
    const v8bf ahi = *(const v8bf*)(Arow + k0 + 8 + kbase);
    v16bf a;
    #pragma unroll
    for (int i = 0; i < 8; ++i) { a[i] = alo[i]; a[8 + i] = ahi[i]; }

    v16bf b0 = *(const v16bf*)(bt + (0 * 16 + (lane & 15)) * BK + bsel);
    v16bf b1 = *(const v16bf*)(bt + (1 * 16 + (lane & 15)) * BK + bsel);
    v16bf b2 = *(const v16bf*)(bt + (2 * 16 + (lane & 15)) * BK + bsel);
    v16bf b3 = *(const v16bf*)(bt + (3 * 16 + (lane & 15)) * BK + bsel);

    acc[0] = __builtin_amdgcn_wmma_f32_16x16x32_bf16(false, a, false, b0, (short)0, acc[0], false, false);
    acc[1] = __builtin_amdgcn_wmma_f32_16x16x32_bf16(false, a, false, b1, (short)0, acc[1], false, false);
    acc[2] = __builtin_amdgcn_wmma_f32_16x16x32_bf16(false, a, false, b2, (short)0, acc[2], false, false);
    acc[3] = __builtin_amdgcn_wmma_f32_16x16x32_bf16(false, a, false, b3, (short)0, acc[3], false, false);
}

// ---------------------------------------------------------------------------
// GEMM1: H = silu(LN1 @ W1 + b1), bf16 out. A:[M][K] bf16, BT:[N][K] bf16.
// Double-buffered async staging: copy tile i+1 while WMMAs consume tile i.
// ---------------------------------------------------------------------------
__global__ void gemm_silu_kernel(const unsigned short* __restrict__ A,
                                 const unsigned short* __restrict__ BT,
                                 const float* __restrict__ bias,
                                 unsigned short* __restrict__ C,
                                 int Ntot, int Ktot) {
    __shared__ unsigned short bt[2][BN * BK];   // 2 x 4 KB staged B tiles
    const int tid = threadIdx.x;
    const int wave = tid >> 5, lane = tid & 31;
    const int m0 = blockIdx.x * BM + wave * 16;
    const int n0 = blockIdx.y * BN;
    const int sr = tid >> 2, sc = (tid & 3) * 8;   // staging coords
    const unsigned short* Arow  = A + (size_t)(m0 + (lane & 15)) * Ktot;
    const unsigned short* Bcol  = BT + (size_t)(n0 + sr) * Ktot + sc;

    v8f acc[4] = {};
    const int nsteps = Ktot / BK;

    async_copy_b128(&bt[0][sr * BK + sc], Bcol);               // prologue tile 0
    wait_async();
    __syncthreads();

    for (int i = 0; i < nsteps; ++i) {
        const int k0 = i * BK;
        if (i + 1 < nsteps) {
            async_copy_b128(&bt[(i + 1) & 1][sr * BK + sc], Bcol + k0 + BK);
            __builtin_prefetch(Arow + k0 + BK, 0, 1);
        }
        wmma_step(Arow, k0, bt[i & 1], lane, acc);             // overlap with DMA
        if (i + 1 < nsteps) {
            wait_async();          // own async stores to next buffer done
            __syncthreads();       // all staging visible; all reads of cur done
        }
    }

    const int ncol = lane & 15;
    const int rbase = (lane < 16) ? 0 : 8;
    for (int nt = 0; nt < 4; ++nt) {
        const int n = n0 + nt * 16 + ncol;
        const float bv = bias[n];
        #pragma unroll
        for (int r = 0; r < 8; ++r) {
            float v = acc[nt][r] + bv;
            v = v / (1.0f + __expf(-v));    // SiLU
            C[(size_t)(m0 + rbase + r) * Ntot + n] = f32_to_bf16_bits(v);
        }
    }
}

// ---------------------------------------------------------------------------
// GEMM2: x2 = x + H @ W2 + b2, f32 out.
// ---------------------------------------------------------------------------
__global__ void gemm_res_kernel(const unsigned short* __restrict__ A,
                                const unsigned short* __restrict__ BT,
                                const float* __restrict__ bias,
                                const float* __restrict__ Xres,
                                float* __restrict__ C,
                                int Ntot, int Ktot) {
    __shared__ unsigned short bt[2][BN * BK];
    const int tid = threadIdx.x;
    const int wave = tid >> 5, lane = tid & 31;
    const int m0 = blockIdx.x * BM + wave * 16;
    const int n0 = blockIdx.y * BN;
    const int sr = tid >> 2, sc = (tid & 3) * 8;
    const unsigned short* Arow = A + (size_t)(m0 + (lane & 15)) * Ktot;
    const unsigned short* Bcol = BT + (size_t)(n0 + sr) * Ktot + sc;

    v8f acc[4] = {};
    const int nsteps = Ktot / BK;

    async_copy_b128(&bt[0][sr * BK + sc], Bcol);
    wait_async();
    __syncthreads();

    for (int i = 0; i < nsteps; ++i) {
        const int k0 = i * BK;
        if (i + 1 < nsteps) {
            async_copy_b128(&bt[(i + 1) & 1][sr * BK + sc], Bcol + k0 + BK);
            __builtin_prefetch(Arow + k0 + BK, 0, 1);
        }
        wmma_step(Arow, k0, bt[i & 1], lane, acc);
        if (i + 1 < nsteps) {
            wait_async();
            __syncthreads();
        }
    }

    const int ncol = lane & 15;
    const int rbase = (lane < 16) ? 0 : 8;
    for (int nt = 0; nt < 4; ++nt) {
        const int n = n0 + nt * 16 + ncol;
        const float bv = bias[n];
        #pragma unroll
        for (int r = 0; r < 8; ++r) {
            const size_t idx = (size_t)(m0 + rbase + r) * Ntot + n;
            C[idx] = Xres[idx] + acc[nt][r] + bv;
        }
    }
}

// ---------------------------------------------------------------------------
// Token mixing: out[b,s,e] = x2[b,s,e] + tb[s] + sum_{t<=s} tw[s-t]*ln2[b,t,e]
// Toeplitz A synthesized from LDS-cached bf16 tw; B = ln2T [E][S].
// ---------------------------------------------------------------------------
__global__ void tokenmix_kernel(const unsigned short* __restrict__ LT, // [B][E][S]
                                const float* __restrict__ X2,          // [B][S][E]
                                const float* __restrict__ tw,
                                const float* __restrict__ tb,
                                float* __restrict__ out) {
    __shared__ unsigned short bt[2][BN * BK];
    __shared__ unsigned short tws[SDIM];   // 8 KB bf16 tw cache
    const int tid = threadIdx.x;
    const int wave = tid >> 5, lane = tid & 31;
    const int b  = blockIdx.z;
    const int s0 = blockIdx.x * BM;   // output rows (sequence)
    const int e0 = blockIdx.y * BN;   // output cols (embedding)
    const int sr = tid >> 2, sc = (tid & 3) * 8;

    for (int i = tid; i < SDIM; i += 256) tws[i] = f32_to_bf16_bits(tw[i]);

    const int sw   = s0 + wave * 16;
    const int srow = sw + (lane & 15);
    const int kbase = (lane < 16) ? 0 : 16;
    const int bsel  = (lane < 16) ? 0 : 16;
    const unsigned short* Bcol =
        LT + (size_t)b * EDIM * SDIM + (size_t)(e0 + sr) * SDIM + sc;
    const unsigned short ZB = 0;

    v8f acc[4] = {};
    const int nsteps = (s0 + BM) / BK;   // causal: t >= s0+BM contributes nothing

    async_copy_b128(&bt[0][sr * BK + sc], Bcol);
    wait_async();
    __syncthreads();                     // also covers tws visibility

    for (int i = 0; i < nsteps; ++i) {
        const int t0 = i * BK;
        if (i + 1 < nsteps)
            async_copy_b128(&bt[(i + 1) & 1][sr * BK + sc], Bcol + t0 + BK);

        // Synthesize causal Toeplitz A fragment (16x32 bf16) from cached tw
        v16bf a;
        #pragma unroll
        for (int j = 0; j < 8; ++j) {
            const int t  = t0 + kbase + j;
            const int t2 = t + 8;
            a[j]     = __builtin_bit_cast(__bf16, (t  <= srow) ? tws[srow - t ] : ZB);
            a[8 + j] = __builtin_bit_cast(__bf16, (t2 <= srow) ? tws[srow - t2] : ZB);
        }

        const unsigned short* btc = bt[i & 1];
        v16bf b0 = *(const v16bf*)(btc + (0 * 16 + (lane & 15)) * BK + bsel);
        v16bf b1 = *(const v16bf*)(btc + (1 * 16 + (lane & 15)) * BK + bsel);
        v16bf b2 = *(const v16bf*)(btc + (2 * 16 + (lane & 15)) * BK + bsel);
        v16bf b3 = *(const v16bf*)(btc + (3 * 16 + (lane & 15)) * BK + bsel);

        acc[0] = __builtin_amdgcn_wmma_f32_16x16x32_bf16(false, a, false, b0, (short)0, acc[0], false, false);
        acc[1] = __builtin_amdgcn_wmma_f32_16x16x32_bf16(false, a, false, b1, (short)0, acc[1], false, false);
        acc[2] = __builtin_amdgcn_wmma_f32_16x16x32_bf16(false, a, false, b2, (short)0, acc[2], false, false);
        acc[3] = __builtin_amdgcn_wmma_f32_16x16x32_bf16(false, a, false, b3, (short)0, acc[3], false, false);

        if (i + 1 < nsteps) {
            wait_async();
            __syncthreads();
        }
    }

    const int ncol = lane & 15;
    const int rbase = (lane < 16) ? 0 : 8;
    for (int nt = 0; nt < 4; ++nt) {
        const int e = e0 + nt * 16 + ncol;
        #pragma unroll
        for (int r = 0; r < 8; ++r) {
            const int s = sw + rbase + r;
            const size_t idx = ((size_t)b * SDIM + s) * EDIM + e;
            out[idx] = X2[idx] + tb[s] + acc[nt][r];
        }
    }
}

// ---------------------------------------------------------------------------
// Host-side orchestration
// ---------------------------------------------------------------------------
extern "C" void kernel_launch(void* const* d_in, const int* in_sizes, int n_in,
                              void* d_out, int out_size, void* d_ws, size_t ws_size,
                              hipStream_t stream) {
    const float* x    = (const float*)d_in[0];
    const float* cn_g = (const float*)d_in[1];
    const float* cn_b = (const float*)d_in[2];
    const float* W1   = (const float*)d_in[3];
    const float* b1   = (const float*)d_in[4];
    const float* W2   = (const float*)d_in[5];
    const float* b2   = (const float*)d_in[6];
    const float* tn_g = (const float*)d_in[7];
    const float* tn_b = (const float*)d_in[8];
    const float* tw   = (const float*)d_in[9];
    const float* tb   = (const float*)d_in[10];
    float* out = (float*)d_out;

    // Workspace layout (bf16 regions first, then f32 region)
    unsigned short* ws16 = (unsigned short*)d_ws;
    unsigned short* ln1  = ws16;                                // M*E
    unsigned short* w1t  = ln1  + (size_t)MTOT * EDIM;          // DF*E
    unsigned short* w2t  = w1t  + (size_t)DFDIM * EDIM;         // E*DF
    unsigned short* hbuf = w2t  + (size_t)EDIM * DFDIM;         // M*DF
    unsigned short* ln2t = hbuf + (size_t)MTOT * DFDIM;         // B*E*S (= M*E)
    float*          x2   = (float*)(ln2t + (size_t)MTOT * EDIM);// M*E f32

    const dim3 blk(256);

    // 1. Weight transpose/convert to bf16 [N][K]
    {
        const size_t total = (size_t)EDIM * DFDIM;
        const int g = (int)((total + 255) / 256);
        convT_kernel<<<g, blk, 0, stream>>>(W1, w1t, EDIM, DFDIM);   // W1T [DF][E]
        convT_kernel<<<g, blk, 0, stream>>>(W2, w2t, DFDIM, EDIM);   // W2T [E][DF]
    }

    // 2. LN1(x) -> ln1 bf16 [M][E]
    ln_rows_kernel<<<MTOT, blk, 0, stream>>>(x, cn_g, cn_b, ln1, SDIM, 0);

    // 3. H = silu(ln1 @ W1 + b1) -> hbuf bf16 [M][DF]
    gemm_silu_kernel<<<dim3(MTOT / BM, DFDIM / BN), blk, 0, stream>>>(
        ln1, w1t, b1, hbuf, DFDIM, EDIM);

    // 4. x2 = x + H @ W2 + b2 -> f32 [M][E]
    gemm_res_kernel<<<dim3(MTOT / BM, EDIM / BN), blk, 0, stream>>>(
        hbuf, w2t, b2, x, x2, EDIM, DFDIM);

    // 5. LN2(x2) -> ln2t bf16 transposed [B][E][S]
    ln_rows_kernel<<<MTOT, blk, 0, stream>>>(x2, tn_g, tn_b, ln2t, SDIM, 1);

    // 6. out = x2 + tb + causal-Toeplitz(tw) applied along S
    tokenmix_kernel<<<dim3(SDIM / BM, EDIM / BN, BDIM), blk, 0, stream>>>(
        ln2t, x2, tw, tb, out);
}
